// VectorQuantize_78743930404913
// MI455X (gfx1250) — matrix operand order repaired and checked by reading, not compiled, not run on hardware
//
#include <hip/hip_runtime.h>
#include <hip/hip_bf16.h>

typedef __attribute__((ext_vector_type(16))) __bf16 v16bf;
typedef __attribute__((ext_vector_type(8)))  float  v8f;

#define B_   16
#define T_   512
#define D_   1024
#define H_   4
#define C_   2048
#define HT_  128
#define DECAY_ 0.1f
#define CC_  128   // codes per LDS chunk (128*128*2B = 32 KB per buffer)

__device__ __forceinline__ unsigned short f2bf(float f) {
  union { float f; unsigned u; } v; v.f = f;
  unsigned u = v.u;
  unsigned r = u + 0x7FFFu + ((u >> 16) & 1u);   // round-to-nearest-even
  return (unsigned short)(r >> 16);
}

union BF16Vec { v16bf v; unsigned short s[16]; uint4 q[2]; };

// ---- CDNA5 async global->LDS (ASYNCcnt-tracked, no VGPR round trip) ----
__device__ __forceinline__ void async_lds_b128(unsigned lds_off, const void* gaddr) {
  asm volatile("global_load_async_to_lds_b128 %0, %1, off"
               :: "v"(lds_off), "v"(gaddr) : "memory");
}
__device__ __forceinline__ void async_lds_b32(unsigned lds_off, const void* gaddr) {
  asm volatile("global_load_async_to_lds_b32 %0, %1, off"
               :: "v"(lds_off), "v"(gaddr) : "memory");
}
__device__ __forceinline__ void wait_async0() {
  asm volatile("s_wait_asynccnt 0x0" ::: "memory");
}
__device__ __forceinline__ unsigned lds_addr32(const void* p) {
  return (unsigned)(uintptr_t)p;   // low 32 bits of generic ptr = LDS byte offset
}

// ---- prep: f32 codebook -> bf16 copy + 0.5*||e||^2 per code ----
__global__ void vq_prep_kernel(const float* __restrict__ ke,
                               unsigned short* __restrict__ cb,
                               float* __restrict__ he2) {
  const int row = blockIdx.x;           // h*C_ + c
  const int t   = threadIdx.x;          // 0..127
  const float v = ke[(size_t)row * HT_ + t];
  cb[(size_t)row * HT_ + t] = f2bf(v);
  float sq = v * v;
  #pragma unroll
  for (int o = 16; o > 0; o >>= 1) sq += __shfl_xor(sq, o, 32);
  __shared__ float red[4];
  if ((t & 31) == 0) red[t >> 5] = sq;
  __syncthreads();
  if (t == 0) he2[row] = 0.5f * (red[0] + red[1] + red[2] + red[3]);
}

__global__ void vq_zero_kernel(float* p, int n) {
  int i = blockIdx.x * blockDim.x + threadIdx.x;
  if (i < n) p[i] = 0.0f;
}

// ---- main: async-double-buffered WMMA GEMM + argmax + gather + atomics ----
// grid = (D_/128, B_, H_), block = 256 (8 waves; each wave owns 16 x-columns)
__global__ void VectorQuantize_78743930404913_kernel(
    const float* __restrict__ x, const float* __restrict__ ke,
    const unsigned short* __restrict__ cb, const float* __restrict__ he2,
    float* __restrict__ emb_sum, float* __restrict__ qout,
    float* __restrict__ indout) {
  __shared__ unsigned short cb_lds[2][CC_ * HT_];   // 2 x 32 KB
  __shared__ float e2_lds[2][CC_];

  const int h    = blockIdx.z;
  const int bi   = blockIdx.y;
  const int tid  = threadIdx.x;
  const int lane = tid & 31;
  const int wave = tid >> 5;
  const int n0   = blockIdx.x * 128 + wave * 16;   // base x-column (di)
  const int col  = n0 + (lane & 15);
  const int khalfB = (lane < 16) ? 0 : 16;         // unused; B uses the o1 K octets, like A
  const int o1     = (lane < 16) ? 0 : 8;          // A: K octet base per half-wave

  const size_t xbase = (size_t)bi * (T_ * D_) + (size_t)h * HT_ * D_;

  const unsigned short* cbh  = cb  + (size_t)h * C_ * HT_;
  const float*          he2h = he2 + (size_t)h * C_;

  // Preload B fragments (x tile, 16 cols x K=128), f32 -> bf16 in-register.
  BF16Vec bfrag[4];
  #pragma unroll
  for (int s = 0; s < 4; ++s) {
    #pragma unroll
    for (int e = 0; e < 16; ++e) {
      int k = s * 32 + o1 + (e & 7) + ((e >> 3) << 4);
      bfrag[s].s[e] = f2bf(x[xbase + (size_t)k * D_ + col]);
    }
  }

  // Issue async prefetch of chunk `c0` into LDS buffer `buf`.
  auto prefetch = [&](int buf, int c0) {
    const uint4* src = (const uint4*)(cbh + (size_t)c0 * HT_);
    uint4* dst = (uint4*)cb_lds[buf];
    #pragma unroll
    for (int i = 0; i < 8; ++i)
      async_lds_b128(lds_addr32(&dst[tid + i * 256]), &src[tid + i * 256]);
    if (tid < CC_)
      async_lds_b32(lds_addr32(&e2_lds[buf][tid]), &he2h[c0 + tid]);
  };

  float best = -3.4e38f;
  int   bidx = 0;

  prefetch(0, 0);
  const int NCHUNK = C_ / CC_;   // 16
  for (int c = 0; c < NCHUNK; ++c) {
    const int cur = c & 1;
    const int c0  = c * CC_;
    wait_async0();        // this wave's loads for buf[cur] are in LDS
    __syncthreads();      // everyone's loads visible; everyone done with buf[1-cur]
    if (c + 1 < NCHUNK) prefetch(1 - cur, c0 + CC_);   // overlap DMA with WMMAs

    const unsigned short* cbuf = cb_lds[cur];
    const float*          ebuf = e2_lds[cur];
    #pragma unroll 1
    for (int sub = 0; sub < CC_ / 16; ++sub) {
      const int m16 = sub * 16;
      const int row = m16 + (lane & 15);   // A-matrix row (code index in chunk)
      v8f acc = {0.f,0.f,0.f,0.f,0.f,0.f,0.f,0.f};
      #pragma unroll
      for (int s = 0; s < 4; ++s) {
        BF16Vec a;
        a.q[0] = *(const uint4*)(&cbuf[row * HT_ + s * 32 + o1]);       // K octet 1
        a.q[1] = *(const uint4*)(&cbuf[row * HT_ + s * 32 + o1 + 16]);  // K octet 2
        acc = __builtin_amdgcn_wmma_f32_16x16x32_bf16(
            false, a.v, false, bfrag[s].v, (short)0, acc, false, false);
      }
      const int halfb = (lane < 16) ? 0 : 8;   // C rows M = r + halfb
      #pragma unroll
      for (int r = 0; r < 8; ++r) {
        float sc = acc[r] - ebuf[m16 + halfb + r];   // xe - 0.5*||e||^2
        int   ci = c0 + m16 + halfb + r;
        if (sc > best) { best = sc; bidx = ci; }     // ascending c => first-occurrence
      }
    }
  }

  // Merge the two half-wave row groups for each column (tie: lower index wins).
  {
    float ob = __shfl_xor(best, 16, 32);
    int   oi = __shfl_xor(bidx, 16, 32);
    if (ob > best || (ob == best && oi < bidx)) { best = ob; bidx = oi; }
  }

  // emb_ind: (b, d, h), int indices written as float
  if (lane < 16) indout[((size_t)bi * D_ + col) * H_ + h] = (float)bidx;

  // Gather quantized rows (fp32 codebook) + atomic EMA accumulation (fp32 x).
  for (int j = 0; j < 16; ++j) {
    int idx_j = __shfl(bidx, j, 32);
    const float4* ep = (const float4*)(ke + ((size_t)h * C_ + idx_j) * HT_);
    float4 ev = ep[lane];
    float4* qp = (float4*)(qout + ((size_t)bi * D_ + (n0 + j)) * (H_ * HT_) + h * HT_);
    qp[lane] = ev;
    float* esrow = emb_sum + ((size_t)h * C_ + idx_j) * HT_;
    #pragma unroll
    for (int kk = lane; kk < HT_; kk += 32) {
      float xv = x[xbase + (size_t)kk * D_ + (n0 + j)];
      atomicAdd(&esrow[kk], xv);
    }
  }
}

// ---- final: new_key_embed = lerp(key_embed, emb_sum, DECAY) ----
__global__ void vq_lerp_kernel(const float* __restrict__ ke,
                               const float* __restrict__ es,
                               const int* __restrict__ key_optim,
                               float* __restrict__ out, int n) {
  int i = blockIdx.x * blockDim.x + threadIdx.x;
  if (i >= n) return;
  float k0 = ke[i];
  out[i] = key_optim[0] ? (k0 + DECAY_ * (es[i] - k0)) : k0;
}

extern "C" void kernel_launch(void* const* d_in, const int* in_sizes, int n_in,
                              void* d_out, int out_size, void* d_ws, size_t ws_size,
                              hipStream_t stream) {
  const float* x         = (const float*)d_in[0];
  const float* ke        = (const float*)d_in[1];
  const int*   key_optim = (const int*)d_in[2];

  // workspace layout: bf16 codebook (2 MB) | half_e2 (32 KB) | emb_sum (4 MB)
  unsigned short* cb  = (unsigned short*)d_ws;
  float* he2 = (float*)((char*)d_ws + (size_t)H_ * C_ * HT_ * 2);
  float* es  = (float*)((char*)d_ws + (size_t)H_ * C_ * HT_ * 2 + (size_t)H_ * C_ * 4);

  // outputs flat in return order: q | emb_ind | new_key_embed
  float* qout   = (float*)d_out;
  float* indout = qout + (size_t)B_ * D_ * H_ * HT_;   // + 8388608
  float* nke    = indout + (size_t)B_ * D_ * H_;       // + 65536

  vq_prep_kernel<<<H_ * C_, HT_, 0, stream>>>(ke, cb, he2);
  vq_zero_kernel<<<(H_ * C_ * HT_ + 255) / 256, 256, 0, stream>>>(es, H_ * C_ * HT_);
  dim3 grid(D_ / 128, B_, H_);
  VectorQuantize_78743930404913_kernel<<<grid, 256, 0, stream>>>(x, ke, cb, he2, es, qout, indout);
  vq_lerp_kernel<<<(H_ * C_ * HT_ + 255) / 256, 256, 0, stream>>>(ke, es, key_optim, nke, H_ * C_ * HT_);
}
